// MiniMaxText01LinearAttention_15144054686029
// MI455X (gfx1250) — matrix operand order repaired and checked by reading, not compiled, not run on hardware
//
#include <hip/hip_runtime.h>
#include <hip/hip_bf16.h>

// ---------------------------------------------------------------------------
// MiniMaxText01 lightning-attention layer for gfx1250 (MI455X, CDNA5).
//
// Compute-bound (~370 GFLOP vs ~10us of HBM traffic at 23.3 TB/s):
//   * big GEMMs  -> v_wmma_f32_16x16x32_bf16, inputs pre-converted to bf16
//                   (one-time pass), tiles staged to LDS by the Tensor Data
//                   Mover (tensor_load_to_lds + s_wait_tensorcnt), double
//                   buffered.  Fragments are pure ds_load_b128 (weights are
//                   pre-transposed to [N][K] so B pairs are contiguous).
//   * attention  -> v_wmma_f32_16x16x4_f32, 128x128 fp32 kv state resident in
//                   the 320KB CDNA5 WGP LDS (one workgroup per head).
// ---------------------------------------------------------------------------

typedef float  v2f   __attribute__((ext_vector_type(2)));
typedef float  v8f   __attribute__((ext_vector_type(8)));
typedef __bf16 v16bf __attribute__((ext_vector_type(16)));
typedef unsigned int u4v   __attribute__((ext_vector_type(4)));
typedef unsigned int u32x4 __attribute__((ext_vector_type(4)));
typedef int          i32x4 __attribute__((ext_vector_type(4)));
typedef int          i32x8 __attribute__((ext_vector_type(8)));

#define N_SEQ 4096
#define HIDD  2048
#define NH    32
#define HD    128
#define INNER 4096
#define QKV_N 12288
#define SBLK  256

#define BM 128
#define BN 128
#define BK 32

#if defined(__has_builtin)
#  if __has_builtin(__builtin_amdgcn_tensor_load_to_lds) && \
      __has_builtin(__builtin_amdgcn_s_wait_tensorcnt)
#    define HAVE_TDM 1
#  endif
#endif
#ifndef HAVE_TDM
#  define HAVE_TDM 0
#endif

__device__ __forceinline__ unsigned short f2bf(float f) {
  unsigned int u = __builtin_bit_cast(unsigned int, f);
  u += 0x7fffu + ((u >> 16) & 1u);            // round-to-nearest-even
  return (unsigned short)(u >> 16);
}

__device__ __forceinline__ float fast_rcp(float x) {
#if defined(__has_builtin) && __has_builtin(__builtin_amdgcn_rcpf)
  return __builtin_amdgcn_rcpf(x);            // v_rcp_f32, no div expansion
#else
  return 1.0f / x;
#endif
}

__device__ __forceinline__ v8f wmma_bf16(v16bf a, v16bf b, v8f c) {
  return __builtin_amdgcn_wmma_f32_16x16x32_bf16(false, a, false, b, (short)0, c,
                                                 false, false);
}
__device__ __forceinline__ v8f wmma_f32(v2f a, v2f b, v8f c) {
  return __builtin_amdgcn_wmma_f32_16x16x4_f32(false, a, false, b, (short)0, c,
                                               false, false);
}

#if HAVE_TDM
// Issue one TDM 2D tile load: bf16 tile (tile_d0 x tile_d1), row stride
// `stride_elems`, from global `gaddr` into LDS offset `lds_off`.
// D# bitfields per CDNA5 ISA ch.8 (group0: count/lds/global/type=2,
// group1: data_size=2B + dims/strides).
__device__ __forceinline__ void tdm_load_2d(unsigned lds_off, const void* gaddr,
                                            int rowlen, int nrows,
                                            int tile_d0, int tile_d1,
                                            long long stride_elems) {
  unsigned long long ga = (unsigned long long)gaddr;
  u32x4 g0;
  g0[0] = 1u;                                           // count=1 (valid)
  g0[1] = lds_off;                                      // lds_addr (bytes)
  g0[2] = (unsigned)(ga & 0xffffffffu);                 // global_addr[31:0]
  g0[3] = (unsigned)((ga >> 32) & 0x01ffffffu)          // global_addr[56:32]
        | 0x80000000u;                                  // type=2 ("image")
  i32x8 g1;
  g1[0] = (1 << 16);                                    // data_size=1 -> 2B
  g1[1] = (rowlen & 0xffff) << 16;                      // tensor_dim0 lo16
  g1[2] = ((rowlen >> 16) & 0xffff) | ((nrows & 0xffff) << 16);
  g1[3] = ((nrows  >> 16) & 0xffff) | ((tile_d0 & 0xffff) << 16);
  g1[4] = (tile_d1 & 0xffff);                           // tile_dim1 (tile_dim2=0)
  g1[5] = (int)(stride_elems & 0xffffffffll);           // dim0_stride[31:0]
  g1[6] = (int)((stride_elems >> 32) & 0xffffll);       // dim0_stride[47:32]
  g1[7] = 0;
  i32x4 z4 = {};
#if defined(__clang_major__) && __clang_major__ >= 23
  i32x8 z8 = {};
  __builtin_amdgcn_tensor_load_to_lds(g0, g1, z4, z4, z8, 0);
#else
  __builtin_amdgcn_tensor_load_to_lds(g0, g1, z4, z4, 0);
#endif
}
#endif

// ---------------------------------------------------------------------------
// One-time precision conversion passes (memory-bound, ~13us total @23.3TB/s).
// ---------------------------------------------------------------------------
__global__ __launch_bounds__(256) void f32_to_bf16_kernel(
    const float* __restrict__ in, unsigned short* __restrict__ out,
    long long n) {
  long long i = (long long)blockIdx.x * 256 + threadIdx.x;
  long long stride = (long long)gridDim.x * 256;
  for (; i < n; i += stride) out[i] = f2bf(in[i]);
}

// W fp32 [K][N] row-major  ->  Wt bf16 [N][K]  (32x32 LDS tile transpose)
__global__ __launch_bounds__(256) void transpose_bf16_kernel(
    const float* __restrict__ W, unsigned short* __restrict__ Wt,
    int K, int N) {
  __shared__ unsigned short tile[32][33];
  const int tx = threadIdx.x & 31, ty = threadIdx.x >> 5;
  const int n0 = blockIdx.x * 32, k0 = blockIdx.y * 32;
#pragma unroll
  for (int i = 0; i < 4; ++i) {
    int k = k0 + ty + i * 8;
    tile[ty + i * 8][tx] = f2bf(W[(size_t)k * N + n0 + tx]);
  }
  __syncthreads();
#pragma unroll
  for (int i = 0; i < 4; ++i) {
    int n = n0 + ty + i * 8;
    Wt[(size_t)n * K + k0 + tx] = tile[tx][ty + i * 8];
  }
}

// ---------------------------------------------------------------------------
// C[M,N] = (silu?)(A @ B^T); A bf16 [M][K], Bt bf16 [N][K] (pre-transposed).
// TDM stages double-buffered LDS tiles; fragments are 2x ds_load_b128 each.
// 8 waves; wave grid 2x4; each wave owns 4x2 tiles of 16x16 (fp32 acc).
// ---------------------------------------------------------------------------
__global__ __launch_bounds__(256) void gemm_bf16_tdm_kernel(
    const unsigned short* __restrict__ A, const unsigned short* __restrict__ Bt,
    float* __restrict__ C, int M, int N, int K, int do_silu)
{
  __shared__ unsigned short As[2][BM][BK];   // bf16 bits, [m][k] packed
  __shared__ unsigned short Bs[2][BN][BK];   // bf16 bits, [n][k] packed

  const int tid  = threadIdx.x;
  const int lane = tid & 31;
  const int wv   = tid >> 5;
  const int ln   = lane & 15;
  const int hi   = lane >> 4;
  const int wm   = (wv >> 2) * 64;
  const int wn   = (wv & 3) * 32;
  const int bm0  = blockIdx.y * BM;
  const int bn0  = blockIdx.x * BN;
  const int nk   = K / BK;

  const v8f vzero = {};
  v8f acc[4][2];
#pragma unroll
  for (int tm = 0; tm < 4; ++tm)
#pragma unroll
    for (int tn = 0; tn < 2; ++tn) acc[tm][tn] = vzero;

  auto issue = [&](int kt, int buf) {
#if HAVE_TDM
    if (tid == 0) {                           // TDM ignores EXEC; wave0 issues
      tdm_load_2d((unsigned)(size_t)&As[buf][0][0],
                  A + (size_t)bm0 * K + kt * BK, K, M, BK, BM, (long long)K);
      tdm_load_2d((unsigned)(size_t)&Bs[buf][0][0],
                  Bt + (size_t)bn0 * K + kt * BK, K, N, BK, BN, (long long)K);
    }
#else
    { // fallback: cooperative vector copy (2 threads/row, 32B each)
      int r = tid >> 1, c0 = (tid & 1) * 16;
      const u4v* sa = (const u4v*)(A + (size_t)(bm0 + r) * K + kt * BK + c0);
      u4v* da = (u4v*)&As[buf][r][c0];
      da[0] = sa[0]; da[1] = sa[1];
      const u4v* sb = (const u4v*)(Bt + (size_t)(bn0 + r) * K + kt * BK + c0);
      u4v* db = (u4v*)&Bs[buf][r][c0];
      db[0] = sb[0]; db[1] = sb[1];
    }
#endif
  };

  issue(0, 0);
  for (int kt = 0; kt < nk; ++kt) {
    const int buf = kt & 1;
    if (kt + 1 < nk) issue(kt + 1, buf ^ 1);  // prefetch next tile pair
#if HAVE_TDM
    if (wv == 0) {                            // TENSORcnt is per-wave
      if (kt + 1 < nk) __builtin_amdgcn_s_wait_tensorcnt(2);
      else             __builtin_amdgcn_s_wait_tensorcnt(0);
    }
#endif
    __syncthreads();

    union { v16bf v; u4v q[2]; } fa[4], fb[2];
#pragma unroll
    for (int tm = 0; tm < 4; ++tm) {          // A frag: 16x32 bf16 layout
      int row = wm + tm * 16 + ln;
      fa[tm].q[0] = *(const u4v*)&As[buf][row][hi * 8];
      fa[tm].q[1] = *(const u4v*)&As[buf][row][16 + hi * 8];
    }
#pragma unroll
    for (int tn = 0; tn < 2; ++tn) {          // B frag: 32x16 bf16 layout
      int col = wn + tn * 16 + ln;
      fb[tn].q[0] = *(const u4v*)&Bs[buf][col][hi * 8];
      fb[tn].q[1] = *(const u4v*)&Bs[buf][col][16 + hi * 8];
    }
#pragma unroll
    for (int tm = 0; tm < 4; ++tm)
#pragma unroll
      for (int tn = 0; tn < 2; ++tn)
        acc[tm][tn] = wmma_bf16(fa[tm].v, fb[tn].v, acc[tm][tn]);
    __syncthreads();
  }

#pragma unroll
  for (int tm = 0; tm < 4; ++tm)
#pragma unroll
    for (int tn = 0; tn < 2; ++tn) {
      int col = bn0 + wn + tn * 16 + ln;
#pragma unroll
      for (int r = 0; r < 8; ++r) {
        int row = bm0 + wm + tm * 16 + r + 8 * hi;
        float x = acc[tm][tn][r];
        if (do_silu) x = x * fast_rcp(1.0f + __expf(-x));
        C[(size_t)row * N + col] = x;
      }
    }
}

// ---------------------------------------------------------------------------
// Lightning attention: one workgroup (8 waves) per head.  fp32 WMMA 16x16x4.
// act: [n][12288] silu(qkv); outh: [n][4096] with out[t][h*128+d].
// ---------------------------------------------------------------------------
__global__ __launch_bounds__(256) void lightning_attn_kernel(
    const float* __restrict__ act, const float* __restrict__ kv0,
    float* __restrict__ outh)
{
  __shared__ float S [HD][HD + 2];          // kv state [d][e]
  __shared__ float Ks[SBLK / 2][HD + 2];    // K chunk  [j][d]
  __shared__ float Vs[SBLK / 2][HD + 2];    // V chunk  [j][e]
  __shared__ float scr[8][16][17];          // per-wave score tile scratch

  const int h    = blockIdx.x;
  const int tid  = threadIdx.x;
  const int lane = tid & 31;
  const int wv   = tid >> 5;
  const int ln   = lane & 15;
  const int hi   = lane >> 4;

  // slope for H=32 (power of two), layer 0: 2^(-0.25*(h+1)) * (1 + 1e-5)
  const float s  = exp2f(-0.25f * (float)(h + 1)) * (1.0f + 1e-5f);
  const float bd = __expf(-s * (float)SBLK);

  for (int i = tid; i < HD * HD; i += 256)
    S[i >> 7][i & 127] = kv0[(size_t)h * HD * HD + i];
  __syncthreads();

  const float* qb = act + h * 384;
  const float* kb = act + h * 384 + 128;
  const float* vb = act + h * 384 + 256;

  const v8f vzero = {};

  for (int blk = 0; blk < N_SEQ / SBLK; ++blk) {
    const int base = blk * SBLK;

    {                                        // stage chunk 0 (rows 0..127)
      int r = tid >> 1, c0 = (tid & 1) * 64;
      const float* kp = kb + (size_t)(base + r) * QKV_N + c0;
      const float* vp = vb + (size_t)(base + r) * QKV_N + c0;
#pragma unroll 8
      for (int i = 0; i < 64; ++i) { Ks[r][c0 + i] = kp[i]; Vs[r][c0 + i] = vp[i]; }
    }
    __syncthreads();

    v8f Sacc[8];                             // Sacc = bd*S (wave's strip)
#pragma unroll
    for (int tn = 0; tn < 8; ++tn)
#pragma unroll
      for (int r = 0; r < 8; ++r)
        Sacc[tn][r] = bd * S[wv * 16 + r + 8 * hi][tn * 16 + ln];

    const int drow = wv * 16 + ln;

    auto kv_update = [&](int jcbase) {       // Sacc += (K*kd)^T @ V
#pragma unroll 1
      for (int jk = 0; jk < 128; jk += 4) {
        int j0 = jk + hi * 2;
        float kd0 = __expf(-s * (float)(SBLK - 1 - (jcbase + j0)));
        float kd1 = __expf(-s * (float)(SBLK - 1 - (jcbase + j0 + 1)));
        v2f a; a.x = Ks[j0][drow] * kd0; a.y = Ks[j0 + 1][drow] * kd1;
#pragma unroll
        for (int tn = 0; tn < 8; ++tn) {
          v2f b; b.x = Vs[j0][tn * 16 + ln]; b.y = Vs[j0 + 1][tn * 16 + ln];
          Sacc[tn] = wmma_f32(a, b, Sacc[tn]);
        }
      }
    };

    auto inter = [&](int mcbase, v8f* acc) { // acc += (Q*qdecay) @ S
      const int ml = mcbase + wv * 16 + ln;
      const float qd = __expf(-s * (float)(ml + 1));
      const float* qr = qb + (size_t)(base + ml) * QKV_N;
#pragma unroll 1
      for (int dk = 0; dk < 128; dk += 4) {
        int d0 = dk + hi * 2;
        v2f a; a.x = qr[d0] * qd; a.y = qr[d0 + 1] * qd;
#pragma unroll
        for (int tn = 0; tn < 8; ++tn) {
          v2f b; b.x = S[d0][tn * 16 + ln]; b.y = S[d0 + 1][tn * 16 + ln];
          acc[tn] = wmma_f32(a, b, acc[tn]);
        }
      }
    };

    auto intra = [&](int mcbase, int jcbase, v8f* acc) {
      const int ml = mcbase + wv * 16 + ln;
      const float* qr = qb + (size_t)(base + ml) * QKV_N;
#pragma unroll 1
      for (int jt = 0; jt < 8; ++jt) {
        v8f sc = vzero;                      // 16x16 score tile: Q @ K^T
#pragma unroll 1
        for (int dk = 0; dk < 128; dk += 4) {
          int d0 = dk + hi * 2;
          v2f a; a.x = qr[d0]; a.y = qr[d0 + 1];
          v2f b; b.x = Ks[jt * 16 + ln][d0]; b.y = Ks[jt * 16 + ln][d0 + 1];
          sc = wmma_f32(a, b, sc);
        }
        const int jcol = jcbase + jt * 16 + ln;
#pragma unroll
        for (int r = 0; r < 8; ++r) {        // causal decay mask
          int mrow = mcbase + wv * 16 + r + 8 * hi;
          float diff = (float)(mrow - jcol);
          float w = (diff >= 0.f) ? __expf(-s * diff) : 0.f;
          scr[wv][r + 8 * hi][ln] = sc[r] * w;
        }
        __builtin_amdgcn_wave_barrier();     // C-layout -> A-layout via LDS
#pragma unroll
        for (int jk = 0; jk < 16; jk += 4) {
          v2f a; a.x = scr[wv][ln][jk + hi * 2];
          a.y = scr[wv][ln][jk + hi * 2 + 1];
#pragma unroll
          for (int tn = 0; tn < 8; ++tn) {
            v2f b;
            b.x = Vs[jt * 16 + jk + hi * 2][tn * 16 + ln];
            b.y = Vs[jt * 16 + jk + hi * 2 + 1][tn * 16 + ln];
            acc[tn] = wmma_f32(a, b, acc[tn]);
          }
        }
        __builtin_amdgcn_wave_barrier();
      }
    };

    auto store_out = [&](int mcbase, v8f* acc) {
#pragma unroll
      for (int tn = 0; tn < 8; ++tn)
#pragma unroll
        for (int r = 0; r < 8; ++r) {
          int t = base + mcbase + wv * 16 + r + 8 * hi;
          outh[(size_t)t * INNER + h * HD + tn * 16 + ln] = acc[tn][r];
        }
    };

    v8f acc[8];

    kv_update(0);

#pragma unroll
    for (int tn = 0; tn < 8; ++tn) acc[tn] = vzero;
    inter(0, acc);
    intra(0, 0, acc);
    store_out(0, acc);

#pragma unroll
    for (int tn = 0; tn < 8; ++tn) acc[tn] = vzero;
    inter(128, acc);
    intra(128, 0, acc);

    __syncthreads();                         // done reading chunk 0 (and S)
    {                                        // stage chunk 1 (rows 128..255)
      int r = tid >> 1, c0 = (tid & 1) * 64;
      const float* kp = kb + (size_t)(base + 128 + r) * QKV_N + c0;
      const float* vp = vb + (size_t)(base + 128 + r) * QKV_N + c0;
#pragma unroll 8
      for (int i = 0; i < 64; ++i) { Ks[r][c0 + i] = kp[i]; Vs[r][c0 + i] = vp[i]; }
    }
    __syncthreads();

    kv_update(128);
    intra(128, 128, acc);
    store_out(128, acc);

    __syncthreads();                         // everyone finished reading S
#pragma unroll
    for (int tn = 0; tn < 8; ++tn)
#pragma unroll
      for (int r = 0; r < 8; ++r)
        S[wv * 16 + r + 8 * hi][tn * 16 + ln] = Sacc[tn][r];
    __syncthreads();                         // S committed for next block
  }
}

// ---------------------------------------------------------------------------
// hid2_bf16 = bf16( sigmoid(gate) * rmsnorm(attn) * w )   (one block per row)
// Emits bf16 directly so the final GEMM consumes it without conversion.
// ---------------------------------------------------------------------------
__global__ __launch_bounds__(256) void norm_gate_kernel(
    const float* __restrict__ attn, const float* __restrict__ gate_in,
    const float* __restrict__ w, unsigned short* __restrict__ out_bf)
{
  __shared__ float red[256];
  const int t   = blockIdx.x;
  const int tid = threadIdx.x;

  float ss = 0.f;
  for (int c = tid; c < INNER; c += 256) {
    float x = attn[(size_t)t * INNER + c];
    ss += x * x;
  }
  red[tid] = ss;
  __syncthreads();
  for (int st = 128; st > 0; st >>= 1) {
    if (tid < st) red[tid] += red[tid + st];
    __syncthreads();
  }
  const float rstd = rsqrtf(red[0] / (float)INNER + 1e-5f);
  for (int c = tid; c < INNER; c += 256) {
    float xn  = attn[(size_t)t * INNER + c] * rstd * w[c];
    float g   = gate_in[(size_t)t * INNER + c];
    float sig = fast_rcp(1.0f + __expf(-g));
    out_bf[(size_t)t * INNER + c] = f2bf(xn * sig);
  }
}

// ---------------------------------------------------------------------------
extern "C" void kernel_launch(void* const* d_in, const int* in_sizes, int n_in,
                              void* d_out, int out_size, void* d_ws, size_t ws_size,
                              hipStream_t stream) {
  const float* hidden = (const float*)d_in[0];
  // d_in[1] = positions (unused by the reference math)
  const float* kv0   = (const float*)d_in[2];
  const float* Wqkv  = (const float*)d_in[3];
  const float* Wgate = (const float*)d_in[4];
  const float* Wout  = (const float*)d_in[5];
  const float* normw = (const float*)d_in[6];
  float* out = (float*)d_out;

  char* p = (char*)d_ws;
  float* act  = (float*)p;  p += (size_t)N_SEQ * QKV_N * 4;   // silu(qkv) fp32
  float* gate = (float*)p;  p += (size_t)N_SEQ * INNER * 4;
  float* attn = (float*)p;  p += (size_t)N_SEQ * INNER * 4;
  unsigned short* hid_bf  = (unsigned short*)p; p += (size_t)N_SEQ * HIDD * 2;
  unsigned short* wqkv_t  = (unsigned short*)p; p += (size_t)QKV_N * HIDD * 2;
  unsigned short* wgate_t = (unsigned short*)p; p += (size_t)INNER * HIDD * 2;
  unsigned short* wout_t  = (unsigned short*)p; p += (size_t)HIDD * INNER * 2;
  unsigned short* hid2_bf = (unsigned short*)p; p += (size_t)N_SEQ * INNER * 2;

  dim3 blk(256);

  // one-time bf16 conversion / weight transposition
  f32_to_bf16_kernel<<<dim3(4096), blk, 0, stream>>>(
      hidden, hid_bf, (long long)N_SEQ * HIDD);
  transpose_bf16_kernel<<<dim3(QKV_N / 32, HIDD / 32), blk, 0, stream>>>(
      Wqkv, wqkv_t, HIDD, QKV_N);
  transpose_bf16_kernel<<<dim3(INNER / 32, HIDD / 32), blk, 0, stream>>>(
      Wgate, wgate_t, HIDD, INNER);
  transpose_bf16_kernel<<<dim3(HIDD / 32, INNER / 32), blk, 0, stream>>>(
      Wout, wout_t, INNER, HIDD);

  // qkv = silu(hidden @ W_qkv)
  gemm_bf16_tdm_kernel<<<dim3(QKV_N / BN, N_SEQ / BM), blk, 0, stream>>>(
      hid_bf, wqkv_t, act, N_SEQ, QKV_N, HIDD, 1);
  // gate = hidden @ W_gate
  gemm_bf16_tdm_kernel<<<dim3(INNER / BN, N_SEQ / BM), blk, 0, stream>>>(
      hid_bf, wgate_t, gate, N_SEQ, INNER, HIDD, 0);
  // lightning attention (one head per workgroup, fp32 WMMA)
  lightning_attn_kernel<<<dim3(NH), blk, 0, stream>>>(act, kv0, attn);
  // hid2 = sigmoid(gate) * rmsnorm(attn)  -> bf16
  norm_gate_kernel<<<dim3(N_SEQ), blk, 0, stream>>>(attn, gate, normw, hid2_bf);
  // out = hid2 @ W_out
  gemm_bf16_tdm_kernel<<<dim3(HIDD / BN, N_SEQ / BM), blk, 0, stream>>>(
      hid2_bf, wout_t, out, N_SEQ, HIDD, INNER, 0);
}
